// Decoder_54786602828568
// MI455X (gfx1250) — matrix operand order repaired and checked by reading, not compiled, not run on hardware
//
#include <hip/hip_runtime.h>

// Problem constants (from reference)
#define T_ 256
#define B_ 128
#define S_ 256
#define H_ 512
#define E_ 512

typedef __bf16 bf16;
typedef __attribute__((ext_vector_type(16))) __bf16 bf16x16;
typedef __attribute__((ext_vector_type(8)))  __bf16 bf16x8;
typedef __attribute__((ext_vector_type(8)))  float  floatx8;

__device__ __forceinline__ unsigned short f2bf(float f) {
    unsigned int u = __float_as_uint(f);
    unsigned int r = u + 0x7fffu + ((u >> 16) & 1u);   // round-to-nearest-even
    return (unsigned short)(r >> 16);
}
__device__ __forceinline__ float sigm(float x) { return 1.f / (1.f + __expf(-x)); }

// ---------------------------------------------------------------------------
// GEMM: D[M=128, N] = A[128, K](bf16, row-major) * W[N, K](bf16, row-major)^T
// Block = 8 waves sharing ONE 16-row A tile (contiguous 16*K bf16 block),
// staged into LDS via global_load_async_to_lds_b128 (ASYNCcnt) once per block.
// Each wave then computes a 16x16 tile: A fragments from LDS (ds_load),
// B fragments streamed from the L2-resident bf16 weights.
// ---------------------------------------------------------------------------
__global__ __launch_bounds__(256) void gemm_bf16_wmma(
    const bf16* __restrict__ A, const bf16* __restrict__ W,
    float* __restrict__ D, int N, int K)
{
    extern __shared__ char smem[];               // 16 * K * 2 bytes
    const int wave = threadIdx.x >> 5;
    const int lane = threadIdx.x & 31;
    const int mt   = blockIdx.x & 7;             // M = 128 -> 8 m-tiles
    const int nt   = (blockIdx.x >> 3) * 8 + wave;

    // ---- stage A tile (rows mt*16 .. mt*16+15, all K) into LDS, async ----
    {
        const char* gbase = (const char*)(A + (size_t)mt * 16 * K);
        unsigned ldsBase = (unsigned)(uintptr_t)smem;     // low 32 bits = LDS offset
        int chunks = (16 * K * 2) >> 4;                   // 16B chunks
        for (int ci = threadIdx.x; ci < chunks; ci += 256) {
            unsigned loff = ldsBase + (ci << 4);
            unsigned long long ga = (unsigned long long)(uintptr_t)(gbase + (ci << 4));
            asm volatile("global_load_async_to_lds_b128 %0, %1, off"
                         :: "v"(loff), "v"(ga) : "memory");
        }
        asm volatile("s_wait_asynccnt 0" ::: "memory");
        __syncthreads();
    }

    const int hi  = lane >> 4;                   // lane half-group
    const int l16 = lane & 15;
    const bf16* arow = (const bf16*)smem + (size_t)l16 * K;   // tile-local row
    const bf16* brow = W + (size_t)(nt * 16 + l16) * K;

    floatx8 acc = {};
    for (int k = 0; k < K; k += 32) {
        // A 16x32 bf16 fragment (from LDS): lanes 0-15 k 0..15, lanes 16-31 k 16..31
        union { bf16x16 v; bf16x8 h[2]; } ua;
        ua.h[0] = *(const bf16x8*)(arow + k + (hi ? 16 : 0));
        ua.h[1] = *(const bf16x8*)(arow + k + (hi ? 24 : 8));
        // B 32x16 bf16 fragment: lane n, lanes 0-15 -> k 0..15, lanes 16-31 -> k 16..31
        bf16x16 bm = *(const bf16x16*)(brow + k + (hi ? 16 : 0));
        acc = __builtin_amdgcn_wmma_f32_16x16x32_bf16(
                  false, ua.v, false, bm, (short)0, acc, false, false);
    }
    // D 16x16 f32: lane col = l16; VGPR r -> row r (+8 for high lane group)
    int col = nt * 16 + l16;
    int mb  = mt * 16 + hi * 8;
#pragma unroll
    for (int r = 0; r < 8; ++r)
        D[(size_t)(mb + r) * N + col] = acc[r];
}

// ---------------------------------------------------------------------------
// Build layer-0 input X0[b, 0:1536] = [ emb[token_t_b] | out_prev | h0_prev ]
// (used only for t = 0; later steps build X0 inside finish_step)
// ---------------------------------------------------------------------------
__global__ __launch_bounds__(256) void build_x0(
    const int* __restrict__ tokens, const float* __restrict__ emb,
    const float* __restrict__ outPrev, const float* __restrict__ h0,
    unsigned short* __restrict__ X0, int t)
{
    int i = blockIdx.x * blockDim.x + threadIdx.x;     // B*1536 threads
    int b = i / 1536, col = i % 1536;
    float v;
    if (col < E_) {
        int tok = tokens[t * B_ + b];
        v = emb[(size_t)tok * E_ + col];
    } else if (col < E_ + H_) {
        v = outPrev[b * H_ + (col - E_)];
    } else {
        v = h0[b * H_ + (col - E_ - H_)];
    }
    X0[i] = f2bf(v);
}

// ---------------------------------------------------------------------------
// LSTM pointwise: gates [B,4H] (+bias), update c,h (fp32); pack h (bf16) into
// up to two GEMM-input slots; optionally also pack an unrelated fp32 source
// (used to pack h1_prev alongside layer-0's update -> saves a kernel).
// ---------------------------------------------------------------------------
__global__ __launch_bounds__(256) void lstm_pw(
    const float* __restrict__ G, const float* __restrict__ bias,
    float* __restrict__ c, float* __restrict__ h,
    unsigned short* __restrict__ p1, int stride1, int col1,
    unsigned short* __restrict__ p2, int stride2, int col2,
    const float* __restrict__ xsrc,
    unsigned short* __restrict__ px, int stridex, int colx)
{
    int i = blockIdx.x * blockDim.x + threadIdx.x;     // B*H threads
    int b = i >> 9, hh = i & 511;
    const float* g = G + (size_t)b * 4 * H_;
    float gi = g[hh]           + bias[hh];
    float gf = g[H_ + hh]      + bias[H_ + hh];
    float gg = g[2 * H_ + hh]  + bias[2 * H_ + hh];
    float go = g[3 * H_ + hh]  + bias[3 * H_ + hh];
    float cn = sigm(gf) * c[i] + sigm(gi) * tanhf(gg);
    float hn = sigm(go) * tanhf(cn);
    c[i] = cn;
    h[i] = hn;
    unsigned short hb = f2bf(hn);
    if (p1) p1[(size_t)b * stride1 + col1 + hh] = hb;
    if (p2) p2[(size_t)b * stride2 + col2 + hh] = hb;
    if (px) px[(size_t)b * stridex + colx + hh] = f2bf(xsrc[i]);
}

// ---------------------------------------------------------------------------
// Attention: scores = ctx[s,b,:]·q[b,:], softmax over S, wc = sum_s a*ctx.
// One block per batch row; q staged in LDS; shared-memory reductions.
// Packs wc (bf16) into Xo[:, 0:H]; writes attn weights at the final step.
// ---------------------------------------------------------------------------
__global__ __launch_bounds__(256) void attn_kernel(
    const float* __restrict__ ctx,      // [S,B,H]
    const float* __restrict__ q,        // [B,H]
    unsigned short* __restrict__ Xo,    // [B,2H] bf16, cols 0..H-1 <- wc
    float* __restrict__ attnOut,        // [B,S] (last-step attention)
    int writeAttn)
{
    int b = blockIdx.x;
    int tid = threadIdx.x;              // 256 == S_
    __shared__ float qsh[H_];
    __shared__ float sc[S_];
    __shared__ float red[256];

    for (int hh = tid; hh < H_; hh += 256) qsh[hh] = q[b * H_ + hh];
    __syncthreads();

    const float* crow = ctx + ((size_t)tid * B_ + b) * H_;
    float s = 0.f;
    for (int hh = 0; hh < H_; hh += 4) {
        float4 cv = *(const float4*)(crow + hh);
        s += cv.x * qsh[hh] + cv.y * qsh[hh + 1] + cv.z * qsh[hh + 2] + cv.w * qsh[hh + 3];
    }
    red[tid] = s;
    __syncthreads();
    for (int off = 128; off > 0; off >>= 1) {
        if (tid < off) red[tid] = fmaxf(red[tid], red[tid + off]);
        __syncthreads();
    }
    float m = red[0];
    __syncthreads();
    float ex = __expf(s - m);
    red[tid] = ex;
    __syncthreads();
    for (int off = 128; off > 0; off >>= 1) {
        if (tid < off) red[tid] += red[tid + off];
        __syncthreads();
    }
    float a = ex / red[0];
    sc[tid] = a;
    __syncthreads();

    if (writeAttn) attnOut[b * S_ + tid] = a;

    for (int hh = tid; hh < H_; hh += 256) {
        float acc = 0.f;
        for (int s2 = 0; s2 < S_; ++s2)
            acc += sc[s2] * ctx[((size_t)s2 * B_ + b) * H_ + hh];
        Xo[(size_t)b * (2 * H_) + hh] = f2bf(acc);
    }
}

// ---------------------------------------------------------------------------
// Step epilogue (fused): out = tanh(D); write outputs[t]; AND build next
// step's X0 = [ emb[token_{t+1}] | out | h0_state ] in bf16.
// ---------------------------------------------------------------------------
__global__ __launch_bounds__(256) void finish_step(
    const float* __restrict__ Dmat,          // [B,H] pre-tanh attn output
    float* __restrict__ outSeq,              // d_out outputs base
    const int* __restrict__ tokens, const float* __restrict__ emb,
    const float* __restrict__ h0,            // h_state layer 0 (post step-t update)
    unsigned short* __restrict__ X0, int t, int buildNext)
{
    int i = blockIdx.x * blockDim.x + threadIdx.x;     // B*1536 threads
    int b = i / 1536, col = i % 1536;
    if (col < E_) {
        if (buildNext) {
            int tok = tokens[(t + 1) * B_ + b];
            X0[i] = f2bf(emb[(size_t)tok * E_ + col]);
        }
    } else if (col < E_ + H_) {
        int j = b * H_ + (col - E_);
        float v = tanhf(Dmat[j]);
        outSeq[(size_t)t * B_ * H_ + j] = v;
        if (buildNext) X0[i] = f2bf(v);
    } else {
        if (buildNext) X0[i] = f2bf(h0[b * H_ + (col - E_ - H_)]);
    }
}

// Concatenate (or just convert) weights to bf16: dst[N, Ka+Kb]
__global__ __launch_bounds__(256) void build_wcat(
    const float* __restrict__ Wa, int Ka,
    const float* __restrict__ Wb, int Kb,
    unsigned short* __restrict__ dst, int total)
{
    int i = blockIdx.x * blockDim.x + threadIdx.x;
    if (i >= total) return;
    int K = Ka + Kb;
    int r = i / K, cc = i % K;
    float v = (cc < Ka) ? Wa[(size_t)r * Ka + cc] : Wb[(size_t)r * Kb + (cc - Ka)];
    dst[i] = f2bf(v);
}

__global__ __launch_bounds__(256) void add_bias(
    const float* __restrict__ a, const float* __restrict__ b,
    float* __restrict__ o, int n)
{
    int i = blockIdx.x * blockDim.x + threadIdx.x;
    if (i < n) o[i] = a[i] + b[i];
}

// ---------------------------------------------------------------------------
extern "C" void kernel_launch(void* const* d_in, const int* in_sizes, int n_in,
                              void* d_out, int out_size, void* d_ws, size_t ws_size,
                              hipStream_t stream)
{
    const int*   tokens = (const int*)  d_in[0];
    const float* h0in   = (const float*)d_in[1];
    const float* c0in   = (const float*)d_in[2];
    const float* ctx    = (const float*)d_in[3];
    const float* init_o = (const float*)d_in[4];
    const float* emb    = (const float*)d_in[5];
    const float* Wih0   = (const float*)d_in[6];
    const float* bih0   = (const float*)d_in[7];
    const float* Whh0   = (const float*)d_in[8];
    const float* bhh0   = (const float*)d_in[9];
    const float* Wih1   = (const float*)d_in[10];
    const float* bih1   = (const float*)d_in[11];
    const float* Whh1   = (const float*)d_in[12];
    const float* bhh1   = (const float*)d_in[13];
    const float* Win    = (const float*)d_in[14];
    const float* Wout   = (const float*)d_in[15];
    float* out = (float*)d_out;

    // Output layout: outputs[T,B,H] | hF[2,B,H] | cF[2,B,H] | attn_last[B,S]
    const size_t O1 = (size_t)T_ * B_ * H_;
    const size_t O2 = O1 + 2ull * B_ * H_;
    const size_t O3 = O2 + 2ull * B_ * H_;

    // Workspace carve-out (256B aligned blocks)
    char* w = (char*)d_ws;
    auto alloc = [&](size_t bytes) { char* p = w; w += (bytes + 255) & ~(size_t)255; return p; };
    unsigned short* Wcat0 = (unsigned short*)alloc(2048ull * 1536 * 2);  // [4H, E+2H]
    unsigned short* Wcat1 = (unsigned short*)alloc(2048ull * 1024 * 2);  // [4H, 2H]
    unsigned short* Wq    = (unsigned short*)alloc(512ull  * 512  * 2);  // [H, H]
    unsigned short* Wo    = (unsigned short*)alloc(512ull  * 1024 * 2);  // [H, 2H]
    float* bias0    = (float*)alloc(2048ull * 4);
    float* bias1    = (float*)alloc(2048ull * 4);
    float* h_state  = (float*)alloc(2ull * B_ * H_ * 4);
    float* c_state  = (float*)alloc(2ull * B_ * H_ * 4);
    unsigned short* X0  = (unsigned short*)alloc((size_t)B_ * 1536 * 2);
    unsigned short* X1  = (unsigned short*)alloc((size_t)B_ * 1024 * 2);
    unsigned short* Xo  = (unsigned short*)alloc((size_t)B_ * 1024 * 2);
    unsigned short* Qin = (unsigned short*)alloc((size_t)B_ * 512  * 2);
    float* G0   = (float*)alloc((size_t)B_ * 2048 * 4);
    float* G1   = (float*)alloc((size_t)B_ * 2048 * 4);
    float* qbuf = (float*)alloc((size_t)B_ * 512 * 4);
    float* Dout = (float*)alloc((size_t)B_ * 512 * 4);

    // One-time prep: bf16 weight concat + fused biases + initial state
    build_wcat<<<(2048 * 1536 + 255) / 256, 256, 0, stream>>>(Wih0, 1024, Whh0, 512, Wcat0, 2048 * 1536);
    build_wcat<<<(2048 * 1024 + 255) / 256, 256, 0, stream>>>(Wih1, 512,  Whh1, 512, Wcat1, 2048 * 1024);
    build_wcat<<<(512 * 512 + 255) / 256,   256, 0, stream>>>(Win,  512,  Win,  0,   Wq,    512 * 512);
    build_wcat<<<(512 * 1024 + 255) / 256,  256, 0, stream>>>(Wout, 1024, Wout, 0,   Wo,    512 * 1024);
    add_bias<<<8, 256, 0, stream>>>(bih0, bhh0, bias0, 2048);
    add_bias<<<8, 256, 0, stream>>>(bih1, bhh1, bias1, 2048);
    hipMemcpyAsync(h_state, h0in, 2ull * B_ * H_ * 4, hipMemcpyDeviceToDevice, stream);
    hipMemcpyAsync(c_state, c0in, 2ull * B_ * H_ * 4, hipMemcpyDeviceToDevice, stream);

    const int pwBlocks = (B_ * H_) / 256;              // 256
    // initial X0 = [emb(token_0) | init_output | h0_initial]
    build_x0<<<(B_ * 1536) / 256, 256, 0, stream>>>(tokens, emb, init_o, h0in, X0, 0);

    for (int t = 0; t < T_; ++t) {
        // layer-0 gates: [128,1536] x [1536,2048]   (LDS = 16*K*2 bytes)
        gemm_bf16_wmma<<<2048 / 16, 256, 16 * 1536 * 2, stream>>>(
            (const bf16*)X0, (const bf16*)Wcat0, G0, 2048, 1536);
        // layer-0 update; pack h0_new -> X1[:, :512] and h1_prev -> X1[:, 512:]
        lstm_pw<<<pwBlocks, 256, 0, stream>>>(G0, bias0, c_state, h_state,
                                              X1, 1024, 0, nullptr, 0, 0,
                                              h_state + (size_t)B_ * H_, X1, 1024, 512);
        // layer-1 gates: [128,1024] x [1024,2048]
        gemm_bf16_wmma<<<2048 / 16, 256, 16 * 1024 * 2, stream>>>(
            (const bf16*)X1, (const bf16*)Wcat1, G1, 2048, 1024);
        // layer-1 update; pack h1_new -> Qin and Xo[:, 512:]
        lstm_pw<<<pwBlocks, 256, 0, stream>>>(G1, bias1,
                                              c_state + (size_t)B_ * H_, h_state + (size_t)B_ * H_,
                                              Qin, 512, 0, Xo, 1024, 512,
                                              nullptr, nullptr, 0, 0);
        // q = h1 @ W_in^T : [128,512] x [512,512]
        gemm_bf16_wmma<<<512 / 16, 256, 16 * 512 * 2, stream>>>(
            (const bf16*)Qin, (const bf16*)Wq, qbuf, 512, 512);
        attn_kernel<<<B_, 256, 0, stream>>>(ctx, qbuf, Xo, out + O3, (t == T_ - 1) ? 1 : 0);
        // attn_out = tanh([wc|h1] @ W_out^T) : [128,1024] x [1024,512]
        gemm_bf16_wmma<<<512 / 16, 256, 16 * 1024 * 2, stream>>>(
            (const bf16*)Xo, (const bf16*)Wo, Dout, 512, 1024);
        // epilogue: tanh + store outputs[t] + build X0 for step t+1
        finish_step<<<(B_ * 1536) / 256, 256, 0, stream>>>(
            Dout, out, tokens, emb, h_state, X0, t, (t + 1 < T_) ? 1 : 0);
    }

    hipMemcpyAsync(out + O1, h_state, 2ull * B_ * H_ * 4, hipMemcpyDeviceToDevice, stream);
    hipMemcpyAsync(out + O2, c_state, 2ull * B_ * H_ * 4, hipMemcpyDeviceToDevice, stream);
}
